// ModulatedDeformConv_1726576855894
// MI455X (gfx1250) — matrix operand order repaired and checked by reading, not compiled, not run on hardware
//
#include <hip/hip_runtime.h>
#include <stdint.h>

// ---- CDNA5 WMMA vector types ----
typedef __attribute__((ext_vector_type(16))) __bf16          v16bf;
typedef __attribute__((ext_vector_type(8)))  float           v8f;
typedef __attribute__((ext_vector_type(16))) unsigned short  v16u;
typedef __attribute__((ext_vector_type(8)))  unsigned short  v8u;

#define NB     8
#define CIN    64
#define HH     128
#define WW     128
#define COUT   64
#define KTAP   9
#define KDIM   (CIN * KTAP)   // 576
#define PTOT   (HH * WW)      // 16384
#define PTILE  128            // one full output row per workgroup
#define KC     32             // WMMA K chunk (bf16 16x16x32)
#define NKC    (KDIM / KC)    // 18

__device__ __forceinline__ unsigned short f2bf(float f) {
  union { float f; uint32_t u; } cvt; cvt.f = f;
  uint32_t u = cvt.u;
  // round-to-nearest-even f32 -> bf16
  uint32_t r = (u + 0x7FFFu + ((u >> 16) & 1u)) >> 16;
  return (unsigned short)r;
}

// Pre-convert weight [Cout=64][Kdim=576] fp32 -> bf16 into workspace (L2-resident).
__global__ void wcvt_kernel(const float* __restrict__ w, unsigned short* __restrict__ wb) {
  int i = blockIdx.x * blockDim.x + threadIdx.x;
  if (i < COUT * KDIM) wb[i] = f2bf(w[i]);
}

struct U16x16 { v8u lo; v8u hi; };  // 32B, bit_cast -> v16bf

__global__ __launch_bounds__(256)
void mdcn_kernel(const float* __restrict__ x,
                 const float* __restrict__ off,
                 const float* __restrict__ msk,
                 const unsigned short* __restrict__ wb,
                 const float* __restrict__ bias,
                 float* __restrict__ out) {
  // Tap table: 9 taps x 128 positions, 4 bilinear weights + 4 gather indices each.
  __shared__ __align__(16) float tw[KTAP * PTILE * 4];   // 18 KB
  __shared__ __align__(16) int   tix[KTAP * PTILE * 4];  // 18 KB
  // Staged B tile (cols chunk), [pos][t] so each lane reads 16 contiguous bf16.
  __shared__ __align__(64) unsigned short Blds[PTILE * KC];  // 8 KB

  const int tid = threadIdx.x;
  const int ho  = blockIdx.x;    // 0..127 : output row == position tile
  const int n   = blockIdx.y;    // batch
  const int p0  = ho * PTILE;    // Wo == PTILE == 128

  // ---------- build tap table (channel-invariant sampling data) ----------
  for (int item = tid; item < KTAP * PTILE; item += 256) {
    int tap = item >> 7;          // 0..8
    int wo  = item & 127;         // position within row
    int p   = p0 + wo;
    const float oy = off[((size_t)(n * 2 * KTAP + 2 * tap)) * PTOT + p];
    const float ox = off[((size_t)(n * 2 * KTAP + 2 * tap + 1)) * PTOT + p];
    const float m  = msk[((size_t)(n * KTAP + tap)) * PTOT + p];
    float ys = oy + (float)(tap / 3) + (float)(ho - 1);   // + by + (ho*stride - pad)
    float xs = ox + (float)(tap % 3) + (float)(wo - 1);
    float y0f = floorf(ys), x0f = floorf(xs);
    float ly = ys - y0f, lx = xs - x0f;
    int y0 = (int)y0f, x0 = (int)x0f;
    int y1 = y0 + 1,  x1 = x0 + 1;
    float vy0 = (y0 >= 0 && y0 < HH) ? 1.f : 0.f;
    float vy1 = (y1 >= 0 && y1 < HH) ? 1.f : 0.f;
    float vx0 = (x0 >= 0 && x0 < WW) ? 1.f : 0.f;
    float vx1 = (x1 >= 0 && x1 < WW) ? 1.f : 0.f;
    int cy0 = min(max(y0, 0), HH - 1), cy1 = min(max(y1, 0), HH - 1);
    int cx0 = min(max(x0, 0), WW - 1), cx1 = min(max(x1, 0), WW - 1);
    int b = item * 4;
    tw[b + 0] = (1.f - ly) * (1.f - lx) * m * vy0 * vx0;
    tw[b + 1] = (1.f - ly) * lx         * m * vy0 * vx1;
    tw[b + 2] = ly * (1.f - lx)         * m * vy1 * vx0;
    tw[b + 3] = ly * lx                 * m * vy1 * vx1;
    tix[b + 0] = cy0 * WW + cx0;
    tix[b + 1] = cy0 * WW + cx1;
    tix[b + 2] = cy1 * WW + cx0;
    tix[b + 3] = cy1 * WW + cx1;
  }
  __syncthreads();

  const int lane = tid & 31;
  const int wv   = tid >> 5;        // 8 waves
  const int mblk = wv & 3;          // Cout block (4 x 16)
  const int ngrp = wv >> 2;         // 0/1 : owns 4 N-blocks = 64 positions
  const int ncol = lane & 15;

  v8f c0 = {}, c1 = {}, c2 = {}, c3 = {};

  const int st = tid & 31;          // staged k-row t (fixed ci/tap per thread per chunk)
  const int pg = tid >> 5;          // 16 positions per thread

  for (int kc = 0; kc < NKC; ++kc) {
    // ---------- stage B tile: cols[kc*32 .. +31][p0 .. p0+127] in bf16 ----------
    {
      int kk  = kc * KC + st;
      int ci  = kk / KTAP;
      int tap = kk - ci * KTAP;
      const float* xp = x + (((size_t)n * CIN + ci) << 14);   // x plane, L2-resident
      #pragma unroll
      for (int i = 0; i < 16; ++i) {
        int pidx = pg * 16 + i;
        int b = (tap * PTILE + pidx) * 4;
        float v = tw[b + 0] * xp[tix[b + 0]]
                + tw[b + 1] * xp[tix[b + 1]]
                + tw[b + 2] * xp[tix[b + 2]]
                + tw[b + 3] * xp[tix[b + 3]];
        Blds[pidx * KC + st] = f2bf(v);
      }
    }
    __syncthreads();

    // ---------- A/B fragments + 4x WMMA ----------
    {
      // A (16-bit 16x32 layout): lane holds row M=lane%16; K = (lane/16)*16 + [0..15]
      // contiguous (matches B).
      int row  = mblk * 16 + (lane & 15);
      int half = lane >> 4;
      const unsigned short* ap = wb + row * KDIM + kc * KC + half * 16;
      U16x16 atmp;
      atmp.lo = *(const v8u*)(ap);
      atmp.hi = *(const v8u*)(ap + 8);
      v16bf a = __builtin_bit_cast(v16bf, atmp);

      // B (32x16): lane holds column N=lane%16, K = (lane/16)*16 + [0..15] contiguous.
      int t0 = half * 16;
      int cb = ngrp * 64 + ncol;
      v16bf b0 = __builtin_bit_cast(v16bf, *(const v16u*)(&Blds[(cb +  0) * KC + t0]));
      v16bf b1 = __builtin_bit_cast(v16bf, *(const v16u*)(&Blds[(cb + 16) * KC + t0]));
      v16bf b2 = __builtin_bit_cast(v16bf, *(const v16u*)(&Blds[(cb + 32) * KC + t0]));
      v16bf b3 = __builtin_bit_cast(v16bf, *(const v16u*)(&Blds[(cb + 48) * KC + t0]));

      c0 = __builtin_amdgcn_wmma_f32_16x16x32_bf16(false, a, false, b0, (short)0, c0, false, false);
      c1 = __builtin_amdgcn_wmma_f32_16x16x32_bf16(false, a, false, b1, (short)0, c1, false, false);
      c2 = __builtin_amdgcn_wmma_f32_16x16x32_bf16(false, a, false, b2, (short)0, c2, false, false);
      c3 = __builtin_amdgcn_wmma_f32_16x16x32_bf16(false, a, false, b3, (short)0, c3, false, false);
    }
    __syncthreads();
  }

  // ---------- epilogue: C layout -> out[n][M][p], + bias ----------
  {
    int mhalf = (lane >> 4) * 8;   // lanes 16-31 hold M = 8 + r
    int cb    = ngrp * 64 + ncol;
    #pragma unroll
    for (int r = 0; r < 8; ++r) {
      int M = mblk * 16 + mhalf + r;
      float bv = bias[M];
      size_t base = (((size_t)n * COUT + M) << 14) + p0;
      out[base + cb +  0] = c0[r] + bv;
      out[base + cb + 16] = c1[r] + bv;
      out[base + cb + 32] = c2[r] + bv;
      out[base + cb + 48] = c3[r] + bv;
    }
  }
}

extern "C" void kernel_launch(void* const* d_in, const int* in_sizes, int n_in,
                              void* d_out, int out_size, void* d_ws, size_t ws_size,
                              hipStream_t stream) {
  const float* x    = (const float*)d_in[0];
  const float* off  = (const float*)d_in[1];
  const float* msk  = (const float*)d_in[2];
  const float* w    = (const float*)d_in[3];
  const float* bias = (const float*)d_in[4];
  unsigned short* wb = (unsigned short*)d_ws;   // 64*576*2 = 73728 B of scratch

  wcvt_kernel<<<(COUT * KDIM + 255) / 256, 256, 0, stream>>>(w, wb);

  dim3 grid(HH, NB);   // 128 output rows x 8 batches
  mdcn_kernel<<<grid, 256, 0, stream>>>(x, off, msk, wb, bias, (float*)d_out);
}
